// ExpressionBert_68616397521045
// MI455X (gfx1250) — matrix operand order repaired and checked by reading, not compiled, not run on hardware
//
#include <hip/hip_runtime.h>
#include <hip/hip_bf16.h>
#include <math.h>

// ---------------------------------------------------------------------------
// ExpressionBert forward for MI455X (gfx1250): bf16 WMMA everywhere, fp32
// accumulation, fp32 master activations for LN/softmax statistics.
// Relative-position attention terms are computed on the WMMA pipe via
//   scores[q,k] = Q.K^T + T[q, q-k+c] + U[k, q-k+c],
//   T = Q_tile @ De_band^T,  U = K_chunk @ De_band^T.
// GEMM A-tiles are staged by the Tensor Data Mover (TENSORcnt) with LDS row
// padding; attention tiles use async global->LDS copies (ASYNCcnt).
// ---------------------------------------------------------------------------

typedef __bf16 bf16;
typedef __attribute__((ext_vector_type(16))) __bf16 v16bf;
typedef __attribute__((ext_vector_type(8)))  __bf16 v8bf;
typedef __attribute__((ext_vector_type(8)))  float  v8f;
typedef __attribute__((ext_vector_type(4)))  int    v4i;
typedef __attribute__((ext_vector_type(8)))  int    v8i;
typedef __attribute__((ext_vector_type(4)))  unsigned int v4u;

constexpr int cB = 8, cS = 512, cF = 5, cD = 768, cL = 6, cH = 12, cI = 3072;
constexpr int cDh = 64;
constexpr int cBS = cB * cS;            // 4096 token rows
constexpr int cR  = 2 * 512 - 1;        // 1023 relative positions

#define GLOBAL_AS __attribute__((address_space(1)))
#define LDS_AS    __attribute__((address_space(3)))

#if defined(__HIP_DEVICE_COMPILE__) && __has_builtin(__builtin_amdgcn_global_load_async_to_lds_b128)
#define HAVE_ASYNC 1
#else
#define HAVE_ASYNC 0
#endif

#if defined(__HIP_DEVICE_COMPILE__) && __has_builtin(__builtin_amdgcn_tensor_load_to_lds)
#define HAVE_TDM 1
#else
#define HAVE_TDM 0
#endif

// 16-byte global -> LDS copy; async (ASYNCcnt) on gfx1250 when builtin exists.
__device__ inline void async_cp16(const bf16* gp, bf16* lp) {
#if HAVE_ASYNC
  __builtin_amdgcn_global_load_async_to_lds_b128((GLOBAL_AS v4i*)gp,
                                                 (LDS_AS v4i*)lp, 0, 0);
#else
  *(v8bf*)lp = *(const v8bf*)gp;
#endif
}

__device__ inline void async_wait() {
#if HAVE_ASYNC
#if __has_builtin(__builtin_amdgcn_s_wait_asynccnt)
  __builtin_amdgcn_s_wait_asynccnt(0);
#else
  asm volatile("s_wait_asynccnt 0" ::: "memory");
#endif
#endif
}

__device__ inline unsigned lds_off(const void* p) {
  return (unsigned)(uintptr_t)(LDS_AS const void*)p;
}

// TDM 2D tile load: copies tile_d1 rows of tile_d0 bf16 elements (row stride
// d0_stride elements) from global to LDS, inserting pad_amount(+1) DWORDs of
// LDS padding after each 2<<pad_interval DWORDs (ISA D# groups 0/1, type=2).
__device__ inline void tdm_load_2d(const void* gaddr, unsigned ldsoff,
                                   int tensor_d0, int tensor_d1, int tile_d0,
                                   int tile_d1, int d0_stride, int pad_int,
                                   int pad_amt) {
#if HAVE_TDM
  unsigned long long ga = (unsigned long long)(uintptr_t)gaddr;
  v4u g0;
  g0[0] = 1u;                                         // count=1, user mode
  g0[1] = ldsoff;                                     // lds_addr (bytes)
  g0[2] = (unsigned)(ga & 0xFFFFFFFFu);               // global_addr[31:0]
  g0[3] = (unsigned)((ga >> 32) & 0x01FFFFFFu) | (2u << 30);  // addr[56:32]|type=2
  v8i g1;
  unsigned w0 = (1u << 16);                           // data_size = 1 (2 bytes)
  if (pad_amt >= 0)
    w0 |= (1u << 20) | ((unsigned)pad_int << 22) | ((unsigned)pad_amt << 25);
  g1[0] = (int)w0;
  g1[1] = (int)(((unsigned)tensor_d0 & 0xFFFFu) << 16);              // dim0 lo
  g1[2] = (int)(((unsigned)tensor_d0 >> 16) |
                (((unsigned)tensor_d1 & 0xFFFFu) << 16));            // dim0 hi|dim1 lo
  g1[3] = (int)(((unsigned)tensor_d1 >> 16) | ((unsigned)tile_d0 << 16));
  g1[4] = (int)((unsigned)tile_d1 & 0xFFFFu);         // tile_dim1 (tile_dim2=0)
  g1[5] = d0_stride;                                  // dim0_stride[31:0]
  g1[6] = 0;                                          // stride hi | dim1_stride lo
  g1[7] = 0;
  v4i z4 = {};
#if defined(__clang_major__) && __clang_major__ >= 23
  v8i z8 = {};
  __builtin_amdgcn_tensor_load_to_lds(g0, g1, z4, z4, z8, 0);
#else
  __builtin_amdgcn_tensor_load_to_lds(g0, g1, z4, z4, 0);
#endif
#endif
}

__device__ inline void tdm_wait() {
#if HAVE_TDM
#if __has_builtin(__builtin_amdgcn_s_wait_tensorcnt)
  __builtin_amdgcn_s_wait_tensorcnt(0);
#else
  asm volatile("s_wait_tensorcnt 0" ::: "memory");
#endif
#endif
}

// ---------------------------------------------------------------------------
// WMMA fragment helpers (layouts per CDNA5 ISA 7.12.2, wave32)
// ---------------------------------------------------------------------------
__device__ inline v16bf frag_a(const bf16* p, int stride) {
  int lane = threadIdx.x & 31;
  int m = lane & 15, lh = lane >> 4;
  const bf16* q = p + m * stride + 8 * lh;
  v8bf lo = *(const v8bf*)q;
  v8bf hi = *(const v8bf*)(q + 16);
  v16bf r;
#pragma unroll
  for (int i = 0; i < 8; ++i) { r[i] = lo[i]; r[i + 8] = hi[i]; }
  return r;
}

// B fragment sourced from an LDS tile stored with row index = N, col = K.
__device__ inline v16bf frag_b(const bf16* p, int stride) {
  int lane = threadIdx.x & 31;
  int n = lane & 15, kh = lane >> 4;
  const bf16* q = p + n * stride + 8 * kh;
  v8bf lo = *(const v8bf*)q;
  v8bf hi = *(const v8bf*)(q + 16);
  v16bf r;
#pragma unroll
  for (int i = 0; i < 8; ++i) { r[i] = lo[i]; r[i + 8] = hi[i]; }
  return r;
}

__device__ inline v8f wmma_bf16(v16bf a, v16bf b, v8f c) {
  return __builtin_amdgcn_wmma_f32_16x16x32_bf16(false, a, false, b, (short)0, c,
                                                 false, false);
}

// ---------------------------------------------------------------------------
// fp32 -> bf16 conversion (weights, dist_emb)
// ---------------------------------------------------------------------------
__global__ void k_cvt(const float* __restrict__ in, bf16* __restrict__ out, int n) {
  for (int i = blockIdx.x * blockDim.x + threadIdx.x; i < n;
       i += gridDim.x * blockDim.x)
    out[i] = (bf16)in[i];
}

// ---------------------------------------------------------------------------
// Embedding: h = LN(x @ in_w + in_b + tte).  One block per token row.
// ---------------------------------------------------------------------------
__global__ __launch_bounds__(256) void k_embed(
    const float* __restrict__ x, const float* __restrict__ inw,
    const float* __restrict__ inb, const float* __restrict__ tte,
    const float* __restrict__ g, const float* __restrict__ bta,
    float* __restrict__ hf, bf16* __restrict__ hb) {
  __shared__ float red[256];
  int row = blockIdx.x, tid = threadIdx.x;
  const float* xr = x + (size_t)row * cF;
  float xv[cF];
#pragma unroll
  for (int f = 0; f < cF; ++f) xv[f] = xr[f];
  float vals[3];
#pragma unroll
  for (int i = 0; i < 3; ++i) {
    int c = tid + 256 * i;
    float a = inb[c] + tte[c];
#pragma unroll
    for (int f = 0; f < cF; ++f) a += xv[f] * inw[f * cD + c];
    vals[i] = a;
  }
  red[tid] = vals[0] + vals[1] + vals[2];
  __syncthreads();
  for (int st = 128; st > 0; st >>= 1) {
    if (tid < st) red[tid] += red[tid + st];
    __syncthreads();
  }
  float mean = red[0] * (1.0f / cD);
  __syncthreads();
  float vs = 0.f;
#pragma unroll
  for (int i = 0; i < 3; ++i) { float d = vals[i] - mean; vs += d * d; }
  red[tid] = vs;
  __syncthreads();
  for (int st = 128; st > 0; st >>= 1) {
    if (tid < st) red[tid] += red[tid + st];
    __syncthreads();
  }
  float rstd = rsqrtf(red[0] * (1.0f / cD) + 1e-12f);
#pragma unroll
  for (int i = 0; i < 3; ++i) {
    int c = tid + 256 * i;
    float o = (vals[i] - mean) * rstd * g[c] + bta[c];
    hf[(size_t)row * cD + c] = o;
    hb[(size_t)row * cD + c] = (bf16)o;
  }
}

// ---------------------------------------------------------------------------
// Residual + LayerNorm: out = LN(resid + x).  One block per token row.
// ---------------------------------------------------------------------------
__global__ __launch_bounds__(256) void k_add_ln(
    const float* __restrict__ resid, const float* __restrict__ x,
    const float* __restrict__ g, const float* __restrict__ bta,
    float* __restrict__ outf, bf16* __restrict__ outb) {
  __shared__ float red[256];
  int row = blockIdx.x, tid = threadIdx.x;
  float vals[3];
#pragma unroll
  for (int i = 0; i < 3; ++i) {
    int c = tid + 256 * i;
    vals[i] = resid[(size_t)row * cD + c] + x[(size_t)row * cD + c];
  }
  red[tid] = vals[0] + vals[1] + vals[2];
  __syncthreads();
  for (int st = 128; st > 0; st >>= 1) {
    if (tid < st) red[tid] += red[tid + st];
    __syncthreads();
  }
  float mean = red[0] * (1.0f / cD);
  __syncthreads();
  float vs = 0.f;
#pragma unroll
  for (int i = 0; i < 3; ++i) { float d = vals[i] - mean; vs += d * d; }
  red[tid] = vs;
  __syncthreads();
  for (int st = 128; st > 0; st >>= 1) {
    if (tid < st) red[tid] += red[tid + st];
    __syncthreads();
  }
  float rstd = rsqrtf(red[0] * (1.0f / cD) + 1e-12f);
#pragma unroll
  for (int i = 0; i < 3; ++i) {
    int c = tid + 256 * i;
    float o = (vals[i] - mean) * rstd * g[c] + bta[c];
    outf[(size_t)row * cD + c] = o;
    outb[(size_t)row * cD + c] = (bf16)o;
  }
}

// ---------------------------------------------------------------------------
// bf16 WMMA GEMM: C[M,N] = A[M,K] @ W[K,N] + bias.
// Block tile 128x128, BK=32, LDS double-buffered. A-tile staged by TDM
// (one tensor_load_to_lds per chunk, LDS pad 4 DW per 16 DW -> stride 40),
// B-tile transposed through registers. 8 waves in 2(M) x 4(N); each wave
// computes a 64x32 tile = 8 WMMAs per K chunk.
// Epilogues: 0 = fp32 store, 1 = bf16 [B,H,S,Dh] (QKV), 2 = GELU bf16.
// ---------------------------------------------------------------------------
__global__ __launch_bounds__(256) void k_gemm(
    const bf16* __restrict__ A, const bf16* __restrict__ W,
    const float* __restrict__ bias, int N, int K, int epi,
    float* __restrict__ outf, bf16* __restrict__ outb) {
  __shared__ alignas(16) bf16 As[2][128 * 40];  // [m][k], stride 40 (80B)
  __shared__ alignas(16) bf16 Bs[2][128 * 40];  // [n][k] transposed, stride 40
  int tid = threadIdx.x, lane = tid & 31, w = tid >> 5;
  int wm = w >> 2, wn = w & 3;
  int m0 = blockIdx.y * 128, n0 = blockIdx.x * 128;
  v8f acc[8];
#pragma unroll
  for (int i = 0; i < 8; ++i) acc[i] = (v8f){};

  const int nk = K >> 5;
  // prologue: stage chunk 0
#if HAVE_TDM
  if (w == 0)
    tdm_load_2d(&A[(size_t)m0 * K], lds_off(&As[0][0]), K, 128, 32, 128, K,
                /*pad_int=*/3, /*pad_amt=*/3);
#endif
#pragma unroll
  for (int j = 0; j < 2; ++j) {
    int idx = tid + 256 * j;
#if !HAVE_TDM
    int ar = idx >> 2, as2 = (idx & 3) * 8;
    async_cp16(&A[(size_t)(m0 + ar) * K + as2], &As[0][ar * 40 + as2]);
#endif
    int br = idx >> 4, bs2 = (idx & 15) * 8;
    v8bf bt = *(const v8bf*)&W[(size_t)br * N + n0 + bs2];
#pragma unroll
    for (int e = 0; e < 8; ++e) Bs[0][(bs2 + e) * 40 + br] = bt[e];
  }
  tdm_wait();
  async_wait();
  __syncthreads();

  for (int kc = 0; kc < nk; ++kc) {
    int cur = kc & 1, nxt = cur ^ 1;
    int k0n = (kc + 1) << 5;
    bool more = (kc + 1 < nk);
    v8bf btreg[2];
    if (more) {  // kick off next chunk: TDM/async A, B into regs
#if HAVE_TDM
      if (w == 0)
        tdm_load_2d(&A[(size_t)m0 * K + k0n], lds_off(&As[nxt][0]), K, 128, 32,
                    128, K, 3, 3);
#endif
#pragma unroll
      for (int j = 0; j < 2; ++j) {
        int idx = tid + 256 * j;
#if !HAVE_TDM
        int ar = idx >> 2, as2 = (idx & 3) * 8;
        async_cp16(&A[(size_t)(m0 + ar) * K + k0n + as2],
                   &As[nxt][ar * 40 + as2]);
#endif
        int br = idx >> 4, bs2 = (idx & 15) * 8;
        btreg[j] = *(const v8bf*)&W[(size_t)(k0n + br) * N + n0 + bs2];
      }
    }
    // compute on current buffers: 8 WMMAs
    v16bf af[4], bfr[2];
#pragma unroll
    for (int i = 0; i < 4; ++i)
      af[i] = frag_a(&As[cur][(64 * wm + 16 * i) * 40], 40);
#pragma unroll
    for (int i = 0; i < 2; ++i)
      bfr[i] = frag_b(&Bs[cur][(32 * wn + 16 * i) * 40], 40);
#pragma unroll
    for (int mi = 0; mi < 4; ++mi)
#pragma unroll
      for (int ni = 0; ni < 2; ++ni)
        acc[mi * 2 + ni] = wmma_bf16(af[mi], bfr[ni], acc[mi * 2 + ni]);
    if (more) {  // scatter B regs transposed into next buffer
#pragma unroll
      for (int j = 0; j < 2; ++j) {
        int idx = tid + 256 * j;
        int br = idx >> 4, bs2 = (idx & 15) * 8;
#pragma unroll
        for (int e = 0; e < 8; ++e) Bs[nxt][(bs2 + e) * 40 + br] = btreg[j][e];
      }
    }
    tdm_wait();
    async_wait();
    __syncthreads();
  }

  int lh = lane >> 4, ln = lane & 15;
#pragma unroll
  for (int mi = 0; mi < 4; ++mi) {
#pragma unroll
    for (int ni = 0; ni < 2; ++ni) {
      v8f cc = acc[mi * 2 + ni];
      int n = n0 + 32 * wn + 16 * ni + ln;
      float bv = bias[n];
#pragma unroll
      for (int v2 = 0; v2 < 8; ++v2) {
        int m = m0 + 64 * wm + 16 * mi + v2 + 8 * lh;
        float val = cc[v2] + bv;
        if (epi == 0) {
          outf[(size_t)m * N + n] = val;
        } else if (epi == 1) {  // [B,S,D] -> [B,H,S,Dh]
          int b = m >> 9, s = m & 511, hd = n >> 6, d = n & 63;
          outb[((size_t)(b * cH + hd) * cS + s) * cDh + d] = (bf16)val;
        } else {  // exact GELU
          float gg = 0.5f * val * (1.0f + erff(val * 0.70710678118f));
          outb[(size_t)m * N + n] = (bf16)gg;
        }
      }
    }
  }
}

// ---------------------------------------------------------------------------
// Attention with relative positions, flash-style online softmax, all matmuls
// (QK^T, T, U, PV) on the WMMA pipe.
// Grid: (S/64 q-tiles, B*H). Block: 128 threads (4 waves), wave w: 16 q rows.
// ---------------------------------------------------------------------------
__global__ __launch_bounds__(128) void k_attn(
    const bf16* __restrict__ q, const bf16* __restrict__ k,
    const bf16* __restrict__ v, const bf16* __restrict__ de,
    const float* __restrict__ mask, bf16* __restrict__ ctx) {
  __shared__ alignas(16) bf16 Qs[64 * 72];      // [q][d]
  __shared__ alignas(16) bf16 Ks[32 * 72];      // [key][d]
  __shared__ alignas(16) bf16 Vs[64 * 40];      // [d][key] transposed
  __shared__ alignas(16) bf16 Des[96 * 72];     // rel band [r_local][d]
  __shared__ alignas(16) bf16 Ps[4 * 16 * 40];  // per-wave P tile [16][32]
  __shared__ alignas(16) bf16 Us[32 * 104];     // U[ki][r_local], stride 104
  __shared__ alignas(16) bf16 Ts[4 * 16 * 56];  // per-wave T[qi][r-16w], str 56

  int tid = threadIdx.x, lane = tid & 31, w = tid >> 5;
  int lh = lane >> 4, ln = lane & 15;
  int qt = blockIdx.x, bh = blockIdx.y;
  int b = bh / cH, hno = bh % cH;
  int q0 = qt * 64;
  const size_t base = (size_t)bh * cS * cDh;

  for (int i = tid; i < 64 * 8; i += 128) {
    int row = i >> 3, seg = (i & 7) * 8;
    async_cp16(&q[base + (size_t)(q0 + row) * cDh + seg], &Qs[row * 72 + seg]);
  }
  async_wait();
  __syncthreads();
  // Q fragments are loop-invariant: load once.
  v16bf a0 = frag_a(&Qs[16 * w * 72], 72);
  v16bf a1 = frag_a(&Qs[16 * w * 72 + 32], 72);

  float mrow[8], lrow[8];
#pragma unroll
  for (int i = 0; i < 8; ++i) { mrow[i] = -1e30f; lrow[i] = 0.f; }
  v8f o0 = {}, o1 = {}, o2 = {}, o3 = {};
  const float scale = 0.125f;  // 1/sqrt(64)

  for (int kc = 0; kc < cS / 32; ++kc) {
    int kb = kc * 32;
    for (int i = tid; i < 32 * 8; i += 128) {
      int row = i >> 3, seg = (i & 7) * 8;
      async_cp16(&k[base + (size_t)(kb + row) * cDh + seg], &Ks[row * 72 + seg]);
      v8bf t = *(const v8bf*)&v[base + (size_t)(kb + row) * cDh + seg];
#pragma unroll
      for (int e = 0; e < 8; ++e) Vs[(seg + e) * 40 + row] = t[e];
    }
    int rbase = q0 - kb + 480;  // r_local = qi - ki + 31, qi in [0,64), ki in [0,32)
    for (int i = tid; i < 96 * 8; i += 128) {
      int row = i >> 3, seg = (i & 7) * 8;
      int r = rbase + row;
      r = r < 0 ? 0 : (r > 1022 ? 1022 : r);
      async_cp16(&de[(size_t)r * cDh + seg], &Des[row * 72 + seg]);
    }
    async_wait();
    __syncthreads();

    // --- scores: Q(16x64) @ K^T(64x32): 4 WMMAs ---
    v8f s0 = {}, s1 = {};
    s0 = wmma_bf16(a0, frag_b(&Ks[0], 72), s0);
    s0 = wmma_bf16(a1, frag_b(&Ks[32], 72), s0);
    s1 = wmma_bf16(a0, frag_b(&Ks[16 * 72], 72), s1);
    s1 = wmma_bf16(a1, frag_b(&Ks[16 * 72 + 32], 72), s1);

    // --- T (wave-local): T[qi, j] = Q[qi] . De[16w + j], j in [0,48): 6 WMMAs
#pragma unroll
    for (int nt = 0; nt < 3; ++nt) {
      v8f tacc = {};
      tacc = wmma_bf16(a0, frag_b(&Des[(16 * w + 16 * nt) * 72], 72), tacc);
      tacc = wmma_bf16(a1, frag_b(&Des[(16 * w + 16 * nt) * 72 + 32], 72), tacc);
#pragma unroll
      for (int v2 = 0; v2 < 8; ++v2)
        Ts[(16 * w + v2 + 8 * lh) * 56 + 16 * nt + ln] = (bf16)tacc[v2];
    }
    // --- U (cooperative): U[ki, r] = K[ki] . De[r], 12 tiles over 4 waves ---
#pragma unroll
    for (int i2 = 0; i2 < 3; ++i2) {
      int c = 3 * w + i2;
      int mt = c / 6, nt = c % 6;
      v16bf ka0 = frag_a(&Ks[16 * mt * 72], 72);
      v16bf ka1 = frag_a(&Ks[16 * mt * 72 + 32], 72);
      v8f uacc = {};
      uacc = wmma_bf16(ka0, frag_b(&Des[16 * nt * 72], 72), uacc);
      uacc = wmma_bf16(ka1, frag_b(&Des[16 * nt * 72 + 32], 72), uacc);
#pragma unroll
      for (int v2 = 0; v2 < 8; ++v2)
        Us[(16 * mt + v2 + 8 * lh) * 104 + 16 * nt + ln] = (bf16)uacc[v2];
    }
    __syncthreads();  // U visible to all waves

    float ma0 = (1.f - mask[b * cS + kb + ln]) * -1e9f;
    float ma1 = (1.f - mask[b * cS + kb + 16 + ln]) * -1e9f;

    // --- gather rel terms + online softmax ---
#pragma unroll
    for (int v2 = 0; v2 < 8; ++v2) {
      int rowl = v2 + 8 * lh;             // q row within wave tile
      int rl0 = rowl - ln + 31 + 16 * w;  // r_local for ki = ln
      int rl1 = rl0 - 16;                 // r_local for ki = 16 + ln
      float rel0 = (float)Ts[(16 * w + rowl) * 56 + (rl0 - 16 * w)] +
                   (float)Us[ln * 104 + rl0];
      float rel1 = (float)Ts[(16 * w + rowl) * 56 + (rl1 - 16 * w)] +
                   (float)Us[(16 + ln) * 104 + rl1];
      float x0 = (s0[v2] + rel0) * scale + ma0;
      float x1 = (s1[v2] + rel1) * scale + ma1;
      float mx = fmaxf(x0, x1);
#pragma unroll
      for (int off = 1; off < 16; off <<= 1)
        mx = fmaxf(mx, __shfl_xor(mx, off, 32));
      float mnew = fmaxf(mrow[v2], mx);
      float alpha = __expf(mrow[v2] - mnew);
      float p0 = __expf(x0 - mnew);
      float p1 = __expf(x1 - mnew);
      float rs = p0 + p1;
#pragma unroll
      for (int off = 1; off < 16; off <<= 1) rs += __shfl_xor(rs, off, 32);
      lrow[v2] = lrow[v2] * alpha + rs;
      mrow[v2] = mnew;
      o0[v2] *= alpha; o1[v2] *= alpha; o2[v2] *= alpha; o3[v2] *= alpha;
      Ps[(16 * w + rowl) * 40 + ln] = (bf16)p0;
      Ps[(16 * w + rowl) * 40 + 16 + ln] = (bf16)p1;
    }

    // --- ctx += P(16x32) @ V(32x64): 4 WMMAs ---
    v16bf pa = frag_a(&Ps[16 * w * 40], 40);
    o0 = wmma_bf16(pa, frag_b(&Vs[0 * 40], 40), o0);
    o1 = wmma_bf16(pa, frag_b(&Vs[16 * 40], 40), o1);
    o2 = wmma_bf16(pa, frag_b(&Vs[32 * 40], 40), o2);
    o3 = wmma_bf16(pa, frag_b(&Vs[48 * 40], 40), o3);
    __syncthreads();
  }

#pragma unroll
  for (int v2 = 0; v2 < 8; ++v2) {
    float rinv = 1.0f / lrow[v2];
    int srow = q0 + 16 * w + v2 + 8 * lh;
    size_t rowbase = ((size_t)b * cS + srow) * cD + hno * cDh;
    ctx[rowbase + 0 + ln]  = (bf16)(o0[v2] * rinv);
    ctx[rowbase + 16 + ln] = (bf16)(o1[v2] * rinv);
    ctx[rowbase + 32 + ln] = (bf16)(o2[v2] * rinv);
    ctx[rowbase + 48 + ln] = (bf16)(o3[v2] * rinv);
  }
}

// ---------------------------------------------------------------------------
// Host side
// ---------------------------------------------------------------------------
extern "C" void kernel_launch(void* const* d_in, const int* in_sizes, int n_in,
                              void* d_out, int out_size, void* d_ws, size_t ws_size,
                              hipStream_t stream) {
  (void)in_sizes; (void)n_in; (void)out_size; (void)ws_size;
  const float* x    = (const float*)d_in[0];
  const float* mask = (const float*)d_in[1];
  const float* inw  = (const float*)d_in[2];
  const float* inb  = (const float*)d_in[3];
  const float* tte  = (const float*)d_in[4];
  const float* elng = (const float*)d_in[5];
  const float* elnb = (const float*)d_in[6];
  const float* dist = (const float*)d_in[7];
  const float* wq = (const float*)d_in[8];  const float* bq = (const float*)d_in[9];
  const float* wk = (const float*)d_in[10]; const float* bk = (const float*)d_in[11];
  const float* wv = (const float*)d_in[12]; const float* bv = (const float*)d_in[13];
  const float* wo = (const float*)d_in[14]; const float* bo = (const float*)d_in[15];
  const float* l1g = (const float*)d_in[16]; const float* l1b = (const float*)d_in[17];
  const float* w1 = (const float*)d_in[18]; const float* b1 = (const float*)d_in[19];
  const float* w2 = (const float*)d_in[20]; const float* b2 = (const float*)d_in[21];
  const float* l2g = (const float*)d_in[22]; const float* l2b = (const float*)d_in[23];

  char* base = (char*)d_ws;
  size_t off = 0;
  auto alloc = [&](size_t bytes) -> char* {
    off = (off + 255) & ~(size_t)255;
    char* p = base + off;
    off += bytes;
    return p;
  };
  float* hf  = (float*)alloc((size_t)cBS * cD * 4);
  bf16*  hb  = (bf16*) alloc((size_t)cBS * cD * 2);
  float* h1f = (float*)alloc((size_t)cBS * cD * 4);
  bf16*  h1b = (bf16*) alloc((size_t)cBS * cD * 2);
  float* tmf = (float*)alloc((size_t)cBS * cD * 4);
  bf16*  qb  = (bf16*) alloc((size_t)cBS * cD * 2);
  bf16*  kb  = (bf16*) alloc((size_t)cBS * cD * 2);
  bf16*  vb  = (bf16*) alloc((size_t)cBS * cD * 2);
  bf16*  cxb = (bf16*) alloc((size_t)cBS * cD * 2);
  bf16*  ffb = (bf16*) alloc((size_t)cBS * cI * 2);
  bf16*  wqb = (bf16*) alloc((size_t)cD * cD * 2);
  bf16*  wkb = (bf16*) alloc((size_t)cD * cD * 2);
  bf16*  wvb = (bf16*) alloc((size_t)cD * cD * 2);
  bf16*  wob = (bf16*) alloc((size_t)cD * cD * 2);
  bf16*  w1b = (bf16*) alloc((size_t)cD * cI * 2);
  bf16*  w2b = (bf16*) alloc((size_t)cI * cD * 2);
  bf16*  deb = (bf16*) alloc((size_t)cR * cDh * 2);

  auto cvt = [&](const float* src, bf16* dst, int n) {
    k_cvt<<<1024, 256, 0, stream>>>(src, dst, n);
  };
  auto gemm = [&](const bf16* A, const bf16* W, const float* bias, int N, int K,
                  int epi, float* of, bf16* ob) {
    dim3 g(N / 128, cBS / 128);
    k_gemm<<<g, 256, 0, stream>>>(A, W, bias, N, K, epi, of, ob);
  };

  k_embed<<<cBS, 256, 0, stream>>>(x, inw, inb, tte, elng, elnb, hf, hb);

  for (int l = 0; l < cL; ++l) {
    const size_t wD2 = (size_t)cD * cD, wDI = (size_t)cD * cI;
    cvt(wq + l * wD2, wqb, (int)wD2);
    cvt(wk + l * wD2, wkb, (int)wD2);
    cvt(wv + l * wD2, wvb, (int)wD2);
    cvt(wo + l * wD2, wob, (int)wD2);
    cvt(w1 + l * wDI, w1b, (int)wDI);
    cvt(w2 + l * wDI, w2b, (int)wDI);
    cvt(dist + (size_t)l * cR * cDh, deb, cR * cDh);

    gemm(hb, wqb, bq + l * cD, cD, cD, 1, nullptr, qb);
    gemm(hb, wkb, bk + l * cD, cD, cD, 1, nullptr, kb);
    gemm(hb, wvb, bv + l * cD, cD, cD, 1, nullptr, vb);

    k_attn<<<dim3(cS / 64, cB * cH), 128, 0, stream>>>(qb, kb, vb, deb, mask, cxb);

    gemm(cxb, wob, bo + l * cD, cD, cD, 0, tmf, nullptr);
    k_add_ln<<<cBS, 256, 0, stream>>>(hf, tmf, l1g + l * cD, l1b + l * cD, h1f, h1b);

    gemm(h1b, w1b, b1 + l * cI, cI, cD, 2, nullptr, ffb);
    gemm(ffb, w2b, b2 + l * cD, cD, cI, 0, tmf, nullptr);

    float* outTarget = (l == cL - 1) ? (float*)d_out : hf;
    k_add_ln<<<cBS, 256, 0, stream>>>(h1f, tmf, l2g + l * cD, l2b + l * cD,
                                      outTarget, hb);
  }
}
